// ProbSparseAttention_8280696947077
// MI455X (gfx1250) — matrix operand order repaired and checked by reading, not compiled, not run on hardware
//
#include <hip/hip_runtime.h>
#include <hip/hip_bf16.h>
#include <math.h>

#define B_  2
#define L_  2048
#define D_  1024
#define H_  16
#define DH_ 64
#define U_  40
#define BL_ (B_ * L_)
#define D2_ (2 * D_)

typedef __bf16 bf16_t;
typedef __attribute__((ext_vector_type(16))) __bf16 v16bf;
typedef __attribute__((ext_vector_type(8)))  __bf16 v8bf;
typedef __attribute__((ext_vector_type(8)))  float  v8f;
typedef int v4i __attribute__((vector_size(16)));
typedef __attribute__((address_space(1))) v4i* gv4i_ptr;
typedef __attribute__((address_space(3))) v4i* lv4i_ptr;

// ---------- helpers ----------

__device__ __forceinline__ bf16_t f2bf(float f) {
    union { float f; unsigned u; } in; in.f = f;
    unsigned u = in.u;
    u = u + 0x7fffu + ((u >> 16) & 1u);           // round-to-nearest-even
    union { unsigned short s; bf16_t b; } out;
    out.s = (unsigned short)(u >> 16);
    return out.b;
}

__device__ __forceinline__ v16bf pack16(v8bf lo, v8bf hi) {
    v16bf r;
#pragma unroll
    for (int i = 0; i < 8; ++i) { r[i] = lo[i]; r[i + 8] = hi[i]; }
    return r;
}

// 16-byte global -> LDS async copy (ASYNCcnt path); sync fallback otherwise.
__device__ __forceinline__ void cp_async16(void* lds_dst, const void* g_src) {
#if __has_builtin(__builtin_amdgcn_global_load_async_to_lds_b128)
    __builtin_amdgcn_global_load_async_to_lds_b128(
        (gv4i_ptr)g_src, (lv4i_ptr)lds_dst, 0, 0);
#else
    *(v8bf*)lds_dst = *(const v8bf*)g_src;
#endif
}

template <int N>
__device__ __forceinline__ void wait_async() {
#if __has_builtin(__builtin_amdgcn_global_load_async_to_lds_b128)
#if __has_builtin(__builtin_amdgcn_s_wait_asynccnt)
    __builtin_amdgcn_s_wait_asynccnt(N);
#endif
#endif
}

// ---------- conversion kernels ----------

__global__ __launch_bounds__(256) void cvt_bf16_kernel(
    const float* __restrict__ x, bf16_t* __restrict__ y, int n)
{
    int i = blockIdx.x * 256 + threadIdx.x;
    if (i < n) y[i] = f2bf(x[i]);
}

// W is (Nrows x Kcols) row-major fp32; Wt is (Kcols x Nrows) row-major bf16.
__global__ __launch_bounds__(256) void transpose_cvt_kernel(
    const float* __restrict__ W, bf16_t* __restrict__ Wt, int Nrows, int Kcols)
{
    __shared__ float tile[32][33];
    const int kb = blockIdx.x * 32;
    const int nb = blockIdx.y * 32;
    const int tx = threadIdx.x & 31;
    const int ty = threadIdx.x >> 5;          // 0..7
#pragma unroll
    for (int r = 0; r < 32; r += 8)
        tile[ty + r][tx] = W[(size_t)(nb + ty + r) * Kcols + kb + tx];
    __syncthreads();
#pragma unroll
    for (int r = 0; r < 32; r += 8)
        Wt[(size_t)(kb + ty + r) * Nrows + nb + tx] = f2bf(tile[tx][ty + r]);
}

// ---------- bf16 WMMA GEMM with async-LDS double buffering ----------
// C(MxN, f32) = A(MxK, bf16) * Bt(KxN, bf16) + bias
// Block tile 128(M) x 128(N), K-step 32. 8 waves: 4(M) x 2(N); wave tile 32x64.
// Steady state: branchless async copy of next tile, s_wait_asynccnt, barrier,
// clause of ds_load_b128 fragments, 8 back-to-back WMMAs, barrier.

#define BM_ 128
#define BN_ 128
#define BK_ 32
#define APAD_ 40     // padded row stride (bf16 elems) for A tile rows
#define BPAD_ 136    // padded row stride (bf16 elems) for B tile rows

__global__ __launch_bounds__(256) void gemm_bf16_kernel(
    const bf16_t* __restrict__ A, const bf16_t* __restrict__ Bt,
    const float* __restrict__ bias, float* __restrict__ C,
    int M, int N, int K)
{
    __shared__ bf16_t As[2][BM_ * APAD_];   // 2 x 10 KB
    __shared__ bf16_t Bs[2][BK_ * BPAD_];   // 2 x 8.5 KB

    const int tid  = threadIdx.x;
    const int lane = tid & 31;
    const int wave = tid >> 5;
    const int gm0 = blockIdx.x * BM_;
    const int gn0 = blockIdx.y * BN_;
    const int m0w = (wave >> 1) * 32;       // wave M offset within block tile
    const int n0w = (wave & 1) * 64;        // wave N offset within block tile

    v8f acc[2][4];
#pragma unroll
    for (int i = 0; i < 2; ++i)
#pragma unroll
        for (int j = 0; j < 4; ++j) acc[i][j] = (v8f){};

    // Branchless per-thread copy slots (2 x 16B chunks per matrix per thread).
    const int am  = tid >> 2;               // A row 0..63 (second chunk: +64)
    const int akc = (tid & 3) << 3;         // A col chunk {0,8,16,24}
    const int bk  = tid >> 4;               // B row 0..15 (second chunk: +16)
    const int bnc = (tid & 15) << 3;        // B col chunk {0..120}

    const bf16_t* gA0 = A + (size_t)(gm0 + am) * K + akc;
    const bf16_t* gA1 = A + (size_t)(gm0 + 64 + am) * K + akc;
    const bf16_t* gB0 = Bt + (size_t)bk * N + gn0 + bnc;
    const bf16_t* gB1 = Bt + (size_t)(bk + 16) * N + gn0 + bnc;

    const int lA0 = am * APAD_ + akc;
    const int lA1 = (64 + am) * APAD_ + akc;
    const int lB0 = bk * BPAD_ + bnc;
    const int lB1 = (bk + 16) * BPAD_ + bnc;

    auto copy_tile = [&](int buf, int k0) {
        cp_async16(&As[buf][lA0], gA0 + k0);
        cp_async16(&As[buf][lA1], gA1 + k0);
        cp_async16(&Bs[buf][lB0], gB0 + (size_t)k0 * N);
        cp_async16(&Bs[buf][lB1], gB1 + (size_t)k0 * N);
    };

    // A fragment addressing (16-bit A 16x32): lanes 0-15 row=lane, K-halves {0..7},{16..23};
    // lanes 16-31 row=lane-16, K-halves {8..15},{24..31}.
    const int ar = lane & 15;
    const int ak = (lane >> 4) << 3;
    const int aOff0 = (m0w + ar) * APAD_ + ak;
    const int aOff1 = (m0w + 16 + ar) * APAD_ + ak;
    // B fragment (16-bit B 32x16): N = lane&15, elements = K in same order as A
    const int bOff = ((lane >> 4) << 3) * BPAD_ + n0w + (lane & 15);

    auto compute = [&](int cur) {
        const bf16_t* Ab = &As[cur][0];
        const bf16_t* Bb = &Bs[cur][0];
        v16bf a0 = pack16(*(const v8bf*)(Ab + aOff0), *(const v8bf*)(Ab + aOff0 + 16));
        v16bf a1 = pack16(*(const v8bf*)(Ab + aOff1), *(const v8bf*)(Ab + aOff1 + 16));
        v16bf b0, b1, b2, b3;
#pragma unroll
        for (int s = 0; s < 16; ++s) {
            const int o = bOff + (s + (s & 8)) * BPAD_;
            b0[s] = Bb[o]; b1[s] = Bb[o + 16]; b2[s] = Bb[o + 32]; b3[s] = Bb[o + 48];
        }
        acc[0][0] = __builtin_amdgcn_wmma_f32_16x16x32_bf16(false, a0, false, b0, (short)0, acc[0][0], false, false);
        acc[1][0] = __builtin_amdgcn_wmma_f32_16x16x32_bf16(false, a1, false, b0, (short)0, acc[1][0], false, false);
        acc[0][1] = __builtin_amdgcn_wmma_f32_16x16x32_bf16(false, a0, false, b1, (short)0, acc[0][1], false, false);
        acc[1][1] = __builtin_amdgcn_wmma_f32_16x16x32_bf16(false, a1, false, b1, (short)0, acc[1][1], false, false);
        acc[0][2] = __builtin_amdgcn_wmma_f32_16x16x32_bf16(false, a0, false, b2, (short)0, acc[0][2], false, false);
        acc[1][2] = __builtin_amdgcn_wmma_f32_16x16x32_bf16(false, a1, false, b2, (short)0, acc[1][2], false, false);
        acc[0][3] = __builtin_amdgcn_wmma_f32_16x16x32_bf16(false, a0, false, b3, (short)0, acc[0][3], false, false);
        acc[1][3] = __builtin_amdgcn_wmma_f32_16x16x32_bf16(false, a1, false, b3, (short)0, acc[1][3], false, false);
    };

    const int nsteps = K >> 5;              // K / 32

    copy_tile(0, 0);
    for (int s = 0; s < nsteps - 1; ++s) {          // steady state: branchless body
        copy_tile((s + 1) & 1, (s + 1) << 5);
        wait_async<4>();                            // our 4 just-issued copies may remain
        __syncthreads();                            // everyone's tile `s` resident
        compute(s & 1);
        __syncthreads();                            // done reading before refill
    }
    wait_async<0>();
    __syncthreads();
    compute((nsteps - 1) & 1);

    // C/D layout: VGPR v, lanes 0-15 -> M=v, N=lane; lanes 16-31 -> M=v+8, N=lane-16
    const int cn = lane & 15;
    const int cm = (lane >> 4) << 3;
#pragma unroll
    for (int j = 0; j < 4; ++j) {
        const int col = gn0 + n0w + j * 16 + cn;
        const float bs = bias[col];
#pragma unroll
        for (int v = 0; v < 8; ++v) {
            const int r0 = gm0 + m0w + cm + v;
            C[(size_t)r0 * N + col]        = acc[0][j][v] + bs;
            C[(size_t)(r0 + 16) * N + col] = acc[1][j][v] + bs;
        }
    }
}

// ---------- sparsity measure: M[b,h,l] = max_u(QK) - mean_u(QK) ----------

__global__ __launch_bounds__(256) void mmeasure_kernel(
    const float* __restrict__ Q, const float* __restrict__ KV,
    const int* __restrict__ idxs, float* __restrict__ Mout)
{
    const int w = blockIdx.x * 8 + (threadIdx.x >> 5);   // 0 .. B*H*L-1
    const int lane = threadIdx.x & 31;
    const int l = w % L_;
    const int h = (w / L_) % H_;
    const int b = w / (L_ * H_);

    const float* qrow = Q + (size_t)(b * L_ + l) * D_ + h * DH_;
    const float q0 = qrow[lane];
    const float q1 = qrow[lane + 32];

    float mx = -3.0e38f;
    float sm = 0.0f;
    for (int u = 0; u < U_; ++u) {
        const int j = idxs[l * U_ + u];
        const float* krow = KV + (size_t)(b * L_ + j) * D2_ + h * DH_;
        float p = q0 * krow[lane] + q1 * krow[lane + 32];
#pragma unroll
        for (int m = 16; m; m >>= 1) p += __shfl_xor(p, m, 32);
        mx = fmaxf(mx, p);
        sm += p;
    }
    if (lane == 0) Mout[w] = mx - sm * (1.0f / U_);
}

// ---------- exact top-U selection per (b,h) row ----------

__global__ __launch_bounds__(256) void topk_kernel(
    const float* __restrict__ Mbuf, int* __restrict__ topIdx)
{
    __shared__ float vals[L_];
    __shared__ float rv[256];
    __shared__ int   ri[256];
    const int row = blockIdx.x;                 // b*H + h
    const int tid = threadIdx.x;
    const float* src = Mbuf + (size_t)row * L_;
    for (int i = tid; i < L_; i += 256) vals[i] = src[i];
    __syncthreads();

    for (int it = 0; it < U_; ++it) {
        float bv = -3.0e38f; int bi = L_;
        for (int i = tid; i < L_; i += 256) {
            const float v = vals[i];
            if (v > bv || (v == bv && i < bi)) { bv = v; bi = i; }
        }
        rv[tid] = bv; ri[tid] = bi;
        __syncthreads();
        for (int off = 128; off; off >>= 1) {
            if (tid < off) {
                if (rv[tid + off] > rv[tid] ||
                    (rv[tid + off] == rv[tid] && ri[tid + off] < ri[tid])) {
                    rv[tid] = rv[tid + off]; ri[tid] = ri[tid + off];
                }
            }
            __syncthreads();
        }
        if (tid == 0) { topIdx[row * U_ + it] = ri[0]; vals[ri[0]] = -3.0e38f; }
        __syncthreads();
    }
}

// ---------- mean of V over sequence per (b,h,d) ----------

__global__ __launch_bounds__(64) void meanv_kernel(
    const float* __restrict__ KV, float* __restrict__ meanV)
{
    const int row = blockIdx.x;                 // b*H + h
    const int b = row / H_, h = row % H_;
    const int d = threadIdx.x;                  // 0..63
    float s = 0.0f;
    for (int l = 0; l < L_; ++l)
        s += KV[(size_t)(b * L_ + l) * D2_ + D_ + h * DH_ + d];
    meanV[row * DH_ + d] = s * (1.0f / L_);
}

// ---------- fill context with broadcast mean(V), as bf16 ----------

__global__ __launch_bounds__(256) void fillctx_kernel(
    const float* __restrict__ meanV, bf16_t* __restrict__ ctxb)
{
    const size_t i = (size_t)blockIdx.x * 256 + threadIdx.x;   // over BL*D
    const int col = (int)(i % D_);
    const int h = col >> 6, d = col & 63;
    const int b = (int)((i / D_) / L_);
    ctxb[i] = f2bf(meanV[(b * H_ + h) * DH_ + d]);
}

// ---------- sparse attention for the top-U queries of each (b,h) ----------

__global__ __launch_bounds__(256) void attn_kernel(
    const float* __restrict__ Q, const float* __restrict__ KV,
    const int* __restrict__ topIdx, bf16_t* __restrict__ ctxb)
{
    __shared__ float sc[L_];     // 8 KB scores
    __shared__ float qs[DH_];
    __shared__ float red[256];

    const int tid = threadIdx.x;
    const int u = blockIdx.x % U_;
    const int h = (blockIdx.x / U_) % H_;
    const int b = blockIdx.x / (U_ * H_);
    const int j = topIdx[(b * H_ + h) * U_ + u];

    if (tid < DH_) qs[tid] = Q[(size_t)(b * L_ + j) * D_ + h * DH_ + tid];
    __syncthreads();

    float lmax = -3.0e38f;
    for (int k = tid; k < L_; k += 256) {
        const float* krow = KV + (size_t)(b * L_ + k) * D2_ + h * DH_;
        float s = 0.0f;
#pragma unroll
        for (int d = 0; d < DH_; ++d) s += qs[d] * krow[d];
        s *= 0.125f;
        sc[k] = s;
        lmax = fmaxf(lmax, s);
    }
    red[tid] = lmax; __syncthreads();
    for (int off = 128; off; off >>= 1) {
        if (tid < off) red[tid] = fmaxf(red[tid], red[tid + off]);
        __syncthreads();
    }
    const float mx = red[0];
    __syncthreads();

    float lsum = 0.0f;
    for (int k = tid; k < L_; k += 256) {
        const float e = __expf(sc[k] - mx);
        sc[k] = e;
        lsum += e;
    }
    red[tid] = lsum; __syncthreads();
    for (int off = 128; off; off >>= 1) {
        if (tid < off) red[tid] += red[tid + off];
        __syncthreads();
    }
    const float inv = 1.0f / red[0];
    __syncthreads();

    const int d = tid & 63;
    const int c = tid >> 6;                 // 0..3
    float acc = 0.0f;
    for (int k = c * (L_ / 4); k < (c + 1) * (L_ / 4); ++k)
        acc += sc[k] * KV[(size_t)(b * L_ + k) * D2_ + D_ + h * DH_ + d];
    red[tid] = acc; __syncthreads();
    if (tid < 64) {
        const float r = red[tid] + red[tid + 64] + red[tid + 128] + red[tid + 192];
        ctxb[(size_t)(b * L_ + j) * D_ + h * DH_ + tid] = f2bf(r * inv);
    }
}

// ---------- host side ----------

extern "C" void kernel_launch(void* const* d_in, const int* in_sizes, int n_in,
                              void* d_out, int out_size, void* d_ws, size_t ws_size,
                              hipStream_t stream) {
    const float* x    = (const float*)d_in[0];
    const float* Wq   = (const float*)d_in[1];
    const float* bq   = (const float*)d_in[2];
    const float* Wkv  = (const float*)d_in[3];
    const float* bkv  = (const float*)d_in[4];
    const float* Wout = (const float*)d_in[5];
    const float* bout = (const float*)d_in[6];
    const int*   idxs = (const int*)d_in[7];
    float* out = (float*)d_out;

    char* ws = (char*)d_ws;
    size_t off = 0;
    auto carve = [&](size_t bytes) -> void* {
        void* p = ws + off;
        off += (bytes + 255) & ~(size_t)255;
        return p;
    };

    bf16_t* xb     = (bf16_t*)carve((size_t)BL_ * D_ * 2);
    bf16_t* wq_t   = (bf16_t*)carve((size_t)D_ * D_ * 2);
    bf16_t* wkv_t  = (bf16_t*)carve((size_t)D_ * D2_ * 2);
    bf16_t* wout_t = (bf16_t*)carve((size_t)D_ * D_ * 2);
    float*  Qb     = (float*)carve((size_t)BL_ * D_ * 4);
    float*  KVb    = (float*)carve((size_t)BL_ * D2_ * 4);
    float*  Mb     = (float*)carve((size_t)B_ * H_ * L_ * 4);
    int*    tIdx   = (int*)carve((size_t)B_ * H_ * U_ * 4);
    float*  mV     = (float*)carve((size_t)B_ * H_ * DH_ * 4);
    bf16_t* ctxb   = (bf16_t*)carve((size_t)BL_ * D_ * 2);
    (void)ws_size; (void)in_sizes; (void)n_in; (void)out_size;

    // 1) convert activations to bf16
    cvt_bf16_kernel<<<(BL_ * D_) / 256, 256, 0, stream>>>(x, xb, BL_ * D_);

    // 2) transpose+convert weights (N x K fp32 -> K x N bf16)
    transpose_cvt_kernel<<<dim3(D_ / 32, D_ / 32), 256, 0, stream>>>(Wq, wq_t, D_, D_);
    transpose_cvt_kernel<<<dim3(D_ / 32, D2_ / 32), 256, 0, stream>>>(Wkv, wkv_t, D2_, D_);
    transpose_cvt_kernel<<<dim3(D_ / 32, D_ / 32), 256, 0, stream>>>(Wout, wout_t, D_, D_);

    // 3) Q = x Wq^T + bq ;  KV = x Wkv^T + bkv   (WMMA GEMMs)
    gemm_bf16_kernel<<<dim3(BL_ / BM_, D_ / BN_), 256, 0, stream>>>(xb, wq_t, bq, Qb, BL_, D_, D_);
    gemm_bf16_kernel<<<dim3(BL_ / BM_, D2_ / BN_), 256, 0, stream>>>(xb, wkv_t, bkv, KVb, BL_, D2_, D_);

    // 4) sparsity measure over sampled keys
    mmeasure_kernel<<<(B_ * H_ * L_) / 8, 256, 0, stream>>>(Qb, KVb, idxs, Mb);

    // 5) top-U query selection per (b,h)
    topk_kernel<<<B_ * H_, 256, 0, stream>>>(Mb, tIdx);

    // 6) mean(V) and context fill
    meanv_kernel<<<B_ * H_, 64, 0, stream>>>(KVb, mV);
    fillctx_kernel<<<(BL_ * D_) / 256, 256, 0, stream>>>(mV, ctxb);

    // 7) full attention for top-U queries, scattered into context
    attn_kernel<<<B_ * H_ * U_, 256, 0, stream>>>(Qb, KVb, tIdx, ctxb);

    // 8) output projection (WMMA GEMM) straight into d_out
    gemm_bf16_kernel<<<dim3(BL_ / BM_, D_ / BN_), 256, 0, stream>>>(ctxb, wout_t, bout, out, BL_, D_, D_);
}
